// EPDBaseModel_with_elevation_48000554500265
// MI455X (gfx1250) — matrix operand order repaired and checked by reading, not compiled, not run on hardware
//
#include <hip/hip_runtime.h>
#include <cstdint>

// ---------------------------------------------------------------------------
// CDNA5 (gfx1250) wave32 WMMA GNN pipeline for the EPD flood model.
// All MLP layers except the 64->4 decoder output run on v_wmma_f32_16x16x32_f16 tensor ops.
// Weights are pre-swizzled into B-fragment lane order so every fragment load
// is 2x ds_load_b128.
// ---------------------------------------------------------------------------

typedef __attribute__((ext_vector_type(16))) _Float16 v16h;
typedef __attribute__((ext_vector_type(8)))  float    v8f;

#define HID 64

__device__ __forceinline__ v8f wmma16(v16h a, v16h b, v8f c) {
  // D = A(16x32 f16) * B(32x16 f16) + C(16x16 f32)
  return __builtin_amdgcn_wmma_f32_16x16x32_f16(
      /*neg_a=*/false, a, /*neg_b=*/false, b,
      /*c_mod=*/(short)0, c, /*reuse_a=*/false, /*reuse_b=*/false);
}

// A fragment (16x32, f16). ISA layout: lanes 0-15 hold M=lane, K in {0..7,16..23};
// lanes 16-31 hold M=lane-16, K in {8..15,24..31}. Per-lane K runs are 8-half
// contiguous -> compiler merges to ds_load_b128 pairs.
__device__ __forceinline__ v16h load_a_frag(const _Float16* s, int row_base,
                                            int ld, int kbase, int lane) {
  int M = lane & 15;
  int khalf = (lane < 16) ? 0 : 8;
  const _Float16* rowp = s + (size_t)(row_base + M) * ld + kbase + khalf;
  v16h a;
#pragma unroll
  for (int j = 0; j < 16; ++j) {
    int k = (j & 7) + ((j >> 3) << 4);   // 0..7, then +16
    a[j] = rowp[k];
  }
  return a;
}

// B fragment from pre-swizzled weights: tile (kt,ntile), lane L holds its 16
// halves contiguously -> 2x ds_load_b128.
__device__ __forceinline__ v16h load_b_swz(const _Float16* s, int kt, int ntile,
                                           int lane) {
  const _Float16* p = s + ((size_t)(((kt << 2) + ntile) << 5) + lane) * 16;
  v16h b;
#pragma unroll
  for (int j = 0; j < 16; ++j) b[j] = p[j];
  return b;
}

// C/D layout: reg r, lane L -> M = r + (L<16?0:8), N = L%16.
__device__ __forceinline__ void store_c_f16(v8f acc, _Float16* s, int ld,
                                            int mtile, int ntile, int lane,
                                            const float* __restrict__ bias,
                                            bool relu) {
  int Nc = ntile * 16 + (lane & 15);
  float bv = bias[Nc];
  int mb = mtile * 16 + ((lane < 16) ? 0 : 8);
#pragma unroll
  for (int r = 0; r < 8; ++r) {
    float v = acc[r] + bv;
    if (relu) v = v > 0.f ? v : 0.f;
    s[(size_t)(mb + r) * ld + Nc] = (_Float16)v;
  }
}

// ---------------------------------------------------------------------------
// Weight prep: convert f32 [K][64] row-major into f16 B-fragment swizzled
// order. Rows >= Kvalid are zero (pads We1 45->64).
// dst element index: ((kt*4 + ntile)*32 + lane)*16 + j
//   k = kt*32 + (lane<16?0:8) + (j&7) + (j>>3)*16 ; n = ntile*16 + (lane&15)
// ---------------------------------------------------------------------------
__global__ void swz_b_kernel(const float* __restrict__ src,
                             _Float16* __restrict__ dst, int K, int Kvalid) {
  int i = blockIdx.x * blockDim.x + threadIdx.x;
  if (i >= K * 64) return;
  int j = i & 15;
  int lane = (i >> 4) & 31;
  int tile = i >> 9;
  int kt = tile >> 2, ntile = tile & 3;
  int k = kt * 32 + ((lane < 16) ? 0 : 8) + (j & 7) + ((j >> 3) << 4);
  int n = ntile * 16 + (lane & 15);
  dst[i] = (_Float16)((k < Kvalid) ? src[k * 64 + n] : 0.f);
}

// ---------------------------------------------------------------------------
// Per-step node feature build: nf16[node][0:21] = [h0(8), runoff window(12), elev]
// stored f16 with stride 32 (cols 21..31 zero).
// ---------------------------------------------------------------------------
__global__ void build_nf_kernel(const float* __restrict__ h0,
                                const float* __restrict__ runoff,
                                const float* __restrict__ elev,
                                _Float16* __restrict__ nf, int n, int step) {
  int i = blockIdx.x * blockDim.x + threadIdx.x;
  if (i >= n * 32) return;
  int node = i >> 5, c = i & 31;
  float v = 0.f;
  if (c < 8)       v = h0[node * 8 + c];
  else if (c < 20) v = runoff[node * 24 + step + (c - 8)];
  else if (c == 20) v = elev[node];
  nf[i] = (_Float16)v;
}

// ---------------------------------------------------------------------------
// Edge encoder: mixed[E,45->64] -> MLP(64->64 relu ->64) -> atomic sum by src.
// Block: 256 thr (8 waves) = 32 edges; waves tile 2 Mx4 N.
// ---------------------------------------------------------------------------
__global__ __launch_bounds__(256) void edge_encode_kernel(
    const _Float16* __restrict__ nf, const float* __restrict__ edge_feat,
    const int* __restrict__ src, const int* __restrict__ tgt,
    const _Float16* __restrict__ w1, const float* __restrict__ b1,
    const _Float16* __restrict__ w2, const float* __restrict__ b2,
    float* __restrict__ coded_x, int nE) {
  __shared__ __align__(16) _Float16 s_in[32][64];
  __shared__ __align__(16) _Float16 s_w1[64 * 64];
  __shared__ __align__(16) _Float16 s_w2[64 * 64];
  __shared__ __align__(16) _Float16 s_hid[32][64];
  __shared__ int s_src[32];

  int t = threadIdx.x;
  int e0 = blockIdx.x * 32;
  __builtin_prefetch(edge_feat + (size_t)(e0 + 32) * 3, 0, 1);

  { // weights -> LDS (32-bit copies)
    const uint32_t* a = (const uint32_t*)w1;
    uint32_t* d = (uint32_t*)s_w1;
    for (int i = t; i < 64 * 64 / 2; i += 256) d[i] = a[i];
    a = (const uint32_t*)w2; d = (uint32_t*)s_w2;
    for (int i = t; i < 64 * 64 / 2; i += 256) d[i] = a[i];
  }
  if (t < 32) {
    int ge = e0 + t; if (ge >= nE) ge = nE - 1;
    s_src[t] = src[ge];
  }
  // stage mixed features [32 edges][64] (cols 45..63 zero)
  for (int i = t; i < 32 * 64; i += 256) {
    int e = i >> 6, c = i & 63;
    int ge = e0 + e; if (ge >= nE) ge = nE - 1;
    _Float16 v = (_Float16)0.f;
    if (c < 3)       v = (_Float16)edge_feat[(size_t)ge * 3 + c];
    else if (c < 24) v = nf[(size_t)src[ge] * 32 + (c - 3)];
    else if (c < 45) v = nf[(size_t)tgt[ge] * 32 + (c - 24)];
    s_in[e][c] = v;
  }
  __syncthreads();

  int lane = t & 31, wave = t >> 5;
  int mtile = wave & 1, ntile = wave >> 1;

  v8f acc = {};
#pragma unroll
  for (int kt = 0; kt < 2; ++kt) {
    v16h a = load_a_frag(&s_in[0][0], mtile * 16, 64, kt * 32, lane);
    v16h b = load_b_swz(s_w1, kt, ntile, lane);
    acc = wmma16(a, b, acc);
  }
  store_c_f16(acc, &s_hid[0][0], 64, mtile, ntile, lane, b1, true);
  __syncthreads();

  v8f acc2 = {};
#pragma unroll
  for (int kt = 0; kt < 2; ++kt) {
    v16h a = load_a_frag(&s_hid[0][0], mtile * 16, 64, kt * 32, lane);
    v16h b = load_b_swz(s_w2, kt, ntile, lane);
    acc2 = wmma16(a, b, acc2);
  }
  { // segment_sum by src via global f32 atomics
    int Nc = ntile * 16 + (lane & 15);
    float bv = b2[Nc];
    int mb = mtile * 16 + ((lane < 16) ? 0 : 8);
#pragma unroll
    for (int r = 0; r < 8; ++r) {
      int e = mb + r;
      if (e0 + e < nE)
        atomicAdd(&coded_x[(size_t)s_src[e] * HID + Nc], acc2[r] + bv);
    }
  }
}

// ---------------------------------------------------------------------------
// Message MLP: concat(coded_x[src], coded_x[tgt]) [E,128] -> 64 relu -> 64,
// atomic sum by tgt.
// ---------------------------------------------------------------------------
__global__ __launch_bounds__(256) void message_kernel(
    const float* __restrict__ coded_x, const int* __restrict__ src,
    const int* __restrict__ tgt, const _Float16* __restrict__ w1,
    const float* __restrict__ b1, const _Float16* __restrict__ w2,
    const float* __restrict__ b2, float* __restrict__ aggr, int nE) {
  __shared__ __align__(16) _Float16 s_in[32][128];
  __shared__ __align__(16) _Float16 s_w1[128 * 64];
  __shared__ __align__(16) _Float16 s_w2[64 * 64];
  __shared__ __align__(16) _Float16 s_hid[32][64];
  __shared__ int s_tgt[32];

  int t = threadIdx.x;
  int e0 = blockIdx.x * 32;

  {
    const uint32_t* a = (const uint32_t*)w1;
    uint32_t* d = (uint32_t*)s_w1;
    for (int i = t; i < 128 * 64 / 2; i += 256) d[i] = a[i];
    a = (const uint32_t*)w2; d = (uint32_t*)s_w2;
    for (int i = t; i < 64 * 64 / 2; i += 256) d[i] = a[i];
  }
  if (t < 32) {
    int ge = e0 + t; if (ge >= nE) ge = nE - 1;
    s_tgt[t] = tgt[ge];
  }
  for (int i = t; i < 32 * 128; i += 256) {
    int e = i >> 7, c = i & 127;
    int ge = e0 + e; if (ge >= nE) ge = nE - 1;
    int node = (c < 64) ? src[ge] : tgt[ge];
    s_in[e][c] = (_Float16)coded_x[(size_t)node * HID + (c & 63)];
  }
  __syncthreads();

  int lane = t & 31, wave = t >> 5;
  int mtile = wave & 1, ntile = wave >> 1;

  v8f acc = {};
#pragma unroll
  for (int kt = 0; kt < 4; ++kt) {
    v16h a = load_a_frag(&s_in[0][0], mtile * 16, 128, kt * 32, lane);
    v16h b = load_b_swz(s_w1, kt, ntile, lane);
    acc = wmma16(a, b, acc);
  }
  store_c_f16(acc, &s_hid[0][0], 64, mtile, ntile, lane, b1, true);
  __syncthreads();

  v8f acc2 = {};
#pragma unroll
  for (int kt = 0; kt < 2; ++kt) {
    v16h a = load_a_frag(&s_hid[0][0], mtile * 16, 64, kt * 32, lane);
    v16h b = load_b_swz(s_w2, kt, ntile, lane);
    acc2 = wmma16(a, b, acc2);
  }
  {
    int Nc = ntile * 16 + (lane & 15);
    float bv = b2[Nc];
    int mb = mtile * 16 + ((lane < 16) ? 0 : 8);
#pragma unroll
    for (int r = 0; r < 8; ++r) {
      int e = mb + r;
      if (e0 + e < nE)
        atomicAdd(&aggr[(size_t)s_tgt[e] * HID + Nc], acc2[r] + bv);
    }
  }
}

// ---------------------------------------------------------------------------
// Node update + decode + skip/clip + h0 roll. 32 nodes per block.
// ---------------------------------------------------------------------------
__global__ __launch_bounds__(256) void update_decode_kernel(
    const float* __restrict__ coded_x, const float* __restrict__ aggr,
    const _Float16* __restrict__ wu1, const float* __restrict__ bu1,
    const _Float16* __restrict__ wu2, const float* __restrict__ bu2,
    const _Float16* __restrict__ wd1, const float* __restrict__ bd1,
    const float* __restrict__ wd2, const float* __restrict__ bd2,
    const float* __restrict__ elev, const float* __restrict__ ground,
    const float* __restrict__ h0_cur, float* __restrict__ h0_next,
    float* __restrict__ out, int n, int step) {
  __shared__ __align__(16) _Float16 s_in[32][128];
  __shared__ __align__(16) _Float16 s_wu1[128 * 64];
  __shared__ __align__(16) _Float16 s_wu2[64 * 64];
  __shared__ __align__(16) _Float16 s_wd1[64 * 64];
  __shared__ __align__(16) _Float16 s_hid[32][64];
  __shared__ __align__(16) _Float16 s_proc[32][64];
  __shared__ __align__(16) _Float16 s_dec[32][64];

  int t = threadIdx.x;
  int n0 = blockIdx.x * 32;

  {
    const uint32_t* a = (const uint32_t*)wu1;
    uint32_t* d = (uint32_t*)s_wu1;
    for (int i = t; i < 128 * 64 / 2; i += 256) d[i] = a[i];
    a = (const uint32_t*)wu2; d = (uint32_t*)s_wu2;
    for (int i = t; i < 64 * 64 / 2; i += 256) d[i] = a[i];
    a = (const uint32_t*)wd1; d = (uint32_t*)s_wd1;
    for (int i = t; i < 64 * 64 / 2; i += 256) d[i] = a[i];
  }
  for (int i = t; i < 32 * 128; i += 256) {
    int r = i >> 7, c = i & 127;
    int g = n0 + r; if (g >= n) g = n - 1;
    float v = (c < 64) ? coded_x[(size_t)g * HID + c]
                       : aggr[(size_t)g * HID + (c - 64)];
    s_in[r][c] = (_Float16)v;
  }
  __syncthreads();

  int lane = t & 31, wave = t >> 5;
  int mtile = wave & 1, ntile = wave >> 1;

  v8f acc = {};
#pragma unroll
  for (int kt = 0; kt < 4; ++kt) {
    v16h a = load_a_frag(&s_in[0][0], mtile * 16, 128, kt * 32, lane);
    v16h b = load_b_swz(s_wu1, kt, ntile, lane);
    acc = wmma16(a, b, acc);
  }
  store_c_f16(acc, &s_hid[0][0], 64, mtile, ntile, lane, bu1, true);
  __syncthreads();

  v8f acc2 = {};
#pragma unroll
  for (int kt = 0; kt < 2; ++kt) {
    v16h a = load_a_frag(&s_hid[0][0], mtile * 16, 64, kt * 32, lane);
    v16h b = load_b_swz(s_wu2, kt, ntile, lane);
    acc2 = wmma16(a, b, acc2);
  }
  store_c_f16(acc2, &s_proc[0][0], 64, mtile, ntile, lane, bu2, false);
  __syncthreads();

  v8f acc3 = {};
#pragma unroll
  for (int kt = 0; kt < 2; ++kt) {
    v16h a = load_a_frag(&s_proc[0][0], mtile * 16, 64, kt * 32, lane);
    v16h b = load_b_swz(s_wd1, kt, ntile, lane);
    acc3 = wmma16(a, b, acc3);
  }
  store_c_f16(acc3, &s_dec[0][0], 64, mtile, ntile, lane, bd1, true);
  __syncthreads();

  if (t < 128) {                       // decoder 64->4 (VALU) + skip/clip
    int i = t >> 2, p = t & 3;
    int g = n0 + i;
    if (g < n) {
      float d = bd2[p];
#pragma unroll 8
      for (int k = 0; k < 64; ++k) d += (float)s_dec[i][k] * wd2[k * 4 + p];
      float el = elev[g];
      float prev_y = h0_cur[(size_t)g * 8 + 7] - el;
      float head = 0.5f * d + 0.5f * prev_y + el;
      float clipped = fminf(head, ground[g]);
      float lower = fmaxf(clipped, el);
      out[(size_t)g * 16 + step + p] = lower;
      h0_next[(size_t)g * 8 + 4 + p] = clipped;
    }
  } else {                             // roll history: new[0:4] = old[4:8]
    int q = t - 128;
    int i = q >> 2, j = q & 3;
    int g = n0 + i;
    if (g < n) h0_next[(size_t)g * 8 + j] = h0_cur[(size_t)g * 8 + 4 + j];
  }
}

// ---------------------------------------------------------------------------
extern "C" void kernel_launch(void* const* d_in, const int* in_sizes, int n_in,
                              void* d_out, int out_size, void* d_ws,
                              size_t ws_size, hipStream_t stream) {
  const float* h0        = (const float*)d_in[0];
  const float* runoff    = (const float*)d_in[1];
  const float* elev      = (const float*)d_in[2];
  const float* ground    = (const float*)d_in[3];
  const float* edge_feat = (const float*)d_in[4];
  const int*   ei        = (const int*)d_in[5];
  const float* We1 = (const float*)d_in[7];  const float* be1 = (const float*)d_in[8];
  const float* We2 = (const float*)d_in[9];  const float* be2 = (const float*)d_in[10];
  const float* Wm1 = (const float*)d_in[11]; const float* bm1 = (const float*)d_in[12];
  const float* Wm2 = (const float*)d_in[13]; const float* bm2 = (const float*)d_in[14];
  const float* Wu1 = (const float*)d_in[15]; const float* bu1 = (const float*)d_in[16];
  const float* Wu2 = (const float*)d_in[17]; const float* bu2 = (const float*)d_in[18];
  const float* Wd1 = (const float*)d_in[19]; const float* bd1 = (const float*)d_in[20];
  const float* Wd2 = (const float*)d_in[21]; const float* bd2 = (const float*)d_in[22];

  const int N = in_sizes[0] / 8;
  const int E = in_sizes[4] / 3;
  const int* src = ei;
  const int* tgt = ei + E;

  // scratch layout
  char* ws = (char*)d_ws;
  size_t off = 0;
  auto take = [&](size_t bytes) -> void* {
    void* p = ws + off;
    off = (off + bytes + 255) & ~(size_t)255;
    return p;
  };
  _Float16* nf      = (_Float16*)take((size_t)N * 32 * 2);
  float*    coded_x = (float*)take((size_t)N * HID * 4);
  float*    aggr    = (float*)take((size_t)N * HID * 4);
  float*    h0a     = (float*)take((size_t)N * 8 * 4);
  float*    h0b     = (float*)take((size_t)N * 8 * 4);
  _Float16* w1p  = (_Float16*)take(64 * 64 * 2);
  _Float16* w2p  = (_Float16*)take(64 * 64 * 2);
  _Float16* wm1p = (_Float16*)take(128 * 64 * 2);
  _Float16* wm2p = (_Float16*)take(64 * 64 * 2);
  _Float16* wu1p = (_Float16*)take(128 * 64 * 2);
  _Float16* wu2p = (_Float16*)take(64 * 64 * 2);
  _Float16* wd1p = (_Float16*)take(64 * 64 * 2);
  (void)ws_size; (void)n_in; (void)out_size;

  // weight prep: f32 -> f16 in B-fragment swizzled order (We1 padded 45->64)
  swz_b_kernel<<<16, 256, 0, stream>>>(We1, w1p, 64, 45);
  swz_b_kernel<<<16, 256, 0, stream>>>(We2, w2p, 64, 64);
  swz_b_kernel<<<32, 256, 0, stream>>>(Wm1, wm1p, 128, 128);
  swz_b_kernel<<<16, 256, 0, stream>>>(Wm2, wm2p, 64, 64);
  swz_b_kernel<<<32, 256, 0, stream>>>(Wu1, wu1p, 128, 128);
  swz_b_kernel<<<16, 256, 0, stream>>>(Wu2, wu2p, 64, 64);
  swz_b_kernel<<<16, 256, 0, stream>>>(Wd1, wd1p, 64, 64);

  hipMemcpyAsync(h0a, h0, (size_t)N * 8 * 4, hipMemcpyDeviceToDevice, stream);

  float* cur = h0a;
  float* nxt = h0b;
  const int edgeBlocks = (E + 31) / 32;
  const int nodeBlocks = (N + 31) / 32;
  const int nfBlocks = (N * 32 + 255) / 256;

  for (int step = 0; step < 16; step += 4) {
    build_nf_kernel<<<nfBlocks, 256, 0, stream>>>(cur, runoff, elev, nf, N, step);
    hipMemsetAsync(coded_x, 0, (size_t)N * HID * 4, stream);
    edge_encode_kernel<<<edgeBlocks, 256, 0, stream>>>(
        nf, edge_feat, src, tgt, w1p, be1, w2p, be2, coded_x, E);
    hipMemsetAsync(aggr, 0, (size_t)N * HID * 4, stream);
    message_kernel<<<edgeBlocks, 256, 0, stream>>>(
        coded_x, src, tgt, wm1p, bm1, wm2p, bm2, aggr, E);
    update_decode_kernel<<<nodeBlocks, 256, 0, stream>>>(
        coded_x, aggr, wu1p, bu1, wu2p, bu2, wd1p, bd1, Wd2, bd2,
        elev, ground, cur, nxt, (float*)d_out, N, step);
    float* tmp = cur; cur = nxt; nxt = tmp;
  }
}
